// ConformerBlock_73615739453810
// MI455X (gfx1250) — compile-verified, matrix operand order checked by reading, not run on hardware
//
#include <hip/hip_runtime.h>
#include <cstdint>
#include <cstddef>

// ---- problem constants (match reference) ----
#define B_   8
#define T_   1024
#define D_   512
#define H_   8
#define DH_  64
#define DF_  2048
#define DC_  1024
#define KW_  31
#define M_   (B_*T_)      // 8192 token rows
#define EPS_ 1e-5f

typedef __attribute__((ext_vector_type(16))) _Float16 v16h;
typedef __attribute__((ext_vector_type(8)))  float    v8f;
typedef __attribute__((ext_vector_type(4)))  unsigned int u32x4_t;
typedef __attribute__((ext_vector_type(8)))  int      i32x8_t;
typedef __attribute__((ext_vector_type(4)))  int      i32x4_t;

// ---- CDNA5 Tensor Data Mover detection (device pass only) ----
#if defined(__AMDGCN__)
  #if __has_builtin(__builtin_amdgcn_tensor_load_to_lds) && \
      __has_builtin(__builtin_amdgcn_s_wait_tensorcnt)
    #define HAS_TDM 1
  #else
    #define HAS_TDM 0
  #endif
#else
  #define HAS_TDM 0
#endif

// =====================================================================
// weight conversion
// =====================================================================
__global__ __launch_bounds__(256) void k_f32_to_f16(const float* __restrict__ s,
                                                    _Float16* __restrict__ d, int n) {
  int i = blockIdx.x * 256 + threadIdx.x;
  if (i < n) d[i] = (_Float16)s[i];
}

// src is [R][C] row-major f32; dst is [C][R] row-major f16 (transpose)
__global__ __launch_bounds__(256) void k_transpose_f32_to_f16(const float* __restrict__ s,
                                                              _Float16* __restrict__ d,
                                                              int R, int C) {
  int i = blockIdx.x * 256 + threadIdx.x;
  if (i < R * C) {
    int r = i / C, c = i % C;
    d[(size_t)c * R + r] = (_Float16)s[i];
  }
}

// =====================================================================
// LayerNorm over D_=512, one block (8 waves) per row. Writes f16 and/or f32.
// =====================================================================
__global__ __launch_bounds__(256) void k_layernorm(const float* __restrict__ x,
                                                   const float* __restrict__ w,
                                                   const float* __restrict__ b,
                                                   _Float16* __restrict__ o16,
                                                   float* __restrict__ o32) {
  int row = blockIdx.x;
  const float* xr = x + (size_t)row * D_;
  int tid = threadIdx.x, lane = tid & 31, wv = tid >> 5;
  float s = 0.f, ss = 0.f;
  for (int i = tid; i < D_; i += 256) { float v = xr[i]; s += v; ss += v * v; }
  for (int off = 16; off; off >>= 1) {
    s  += __shfl_down(s,  off, 32);
    ss += __shfl_down(ss, off, 32);
  }
  __shared__ float ps[8], pss[8];
  __shared__ float mean_s, rstd_s;
  if (lane == 0) { ps[wv] = s; pss[wv] = ss; }
  __syncthreads();
  if (tid == 0) {
    float a = 0.f, c = 0.f;
    for (int i = 0; i < 8; ++i) { a += ps[i]; c += pss[i]; }
    float mean = a / (float)D_;
    float var  = c / (float)D_ - mean * mean;
    mean_s = mean;
    rstd_s = rsqrtf(var + EPS_);
  }
  __syncthreads();
  float mean = mean_s, rstd = rstd_s;
  for (int i = tid; i < D_; i += 256) {
    float v = (xr[i] - mean) * rstd * w[i] + b[i];
    if (o16) o16[(size_t)row * D_ + i] = (_Float16)v;
    if (o32) o32[(size_t)row * D_ + i] = v;
  }
}

// =====================================================================
// WMMA GEMM:  C[M,N] = act(A[M,K] @ W^T + bias)  (+ optional residual)
// W (the B operand) is pre-transposed in memory: Bt[N][K] row-major, so
// both LDS tiles are [128 rows][32 halves] and both fragment reads are
// contiguous 32-bit pairs (coalesce to ds_load_b128).
// 128x128 block tile, BK=32, 8 wave32s (4M x 2N), 2x4 WMMA accs per wave.
// Double-buffered LDS; wave0 TDM-stages the A tile, wave1 the B tile;
// all waves s_wait_tensorcnt + barrier.
// =====================================================================
__global__ __launch_bounds__(256) void k_gemm_wmma(
    const _Float16* __restrict__ A, const _Float16* __restrict__ Bt,
    const float* __restrict__ bias,
    float* __restrict__ o32, _Float16* __restrict__ o16,
    const float* __restrict__ res, float resScale,
    int N, int K, int act) {
  __shared__ _Float16 As[2][128][32];   // [buf][m][k]  8 KB each
  __shared__ _Float16 Bs[2][128][32];   // [buf][n][k]  8 KB each

  const int tid  = threadIdx.x;
  const int lane = tid & 31;
  const int wave = tid >> 5;
  const int wm   = wave & 3;   // 4 M-slices of 32
  const int wn   = wave >> 2;  // 2 N-slices of 64
  const int m0 = blockIdx.y * 128, n0 = blockIdx.x * 128;

  v8f acc[2][4];
#pragma unroll
  for (int mt = 0; mt < 2; ++mt)
#pragma unroll
    for (int nt = 0; nt < 4; ++nt)
#pragma unroll
      for (int r = 0; r < 8; ++r) acc[mt][nt][r] = 0.f;

  const int trow = tid >> 1, tcol = (tid & 1) * 16;   // fallback staging map

#if HAS_TDM
  // One TDM 2D-tile descriptor: 32 halves/row x 128 rows, row stride = K
  // elements, 2-byte elements, rows packed contiguously into LDS.
  auto tdm_tile = [&](const _Float16* gsrc, void* ldst, int rowStride, int nrows) {
    uint64_t ga = (uint64_t)(uintptr_t)gsrc;
    uint32_t la = (uint32_t)(uintptr_t)ldst;
    u32x4_t g0 = { 1u,                                    // count=1 (valid user D#)
                   la,                                    // lds_addr
                   (uint32_t)ga,                          // global_addr[31:0]
                   (uint32_t)((ga >> 32) & 0x01FFFFFFu) | 0x80000000u }; // [56:32]|type=2
    i32x8_t g1 = { (int)(1u << 16),                       // data_size=1 -> 2 bytes
                   (int)((unsigned)(rowStride & 0xFFFF) << 16),          // dim0[15:0]
                   (int)(((unsigned)rowStride >> 16) |
                         ((unsigned)(nrows & 0xFFFF) << 16)),            // dim0 hi|dim1 lo
                   (int)(((unsigned)nrows >> 16) | (32u << 16)),         // dim1 hi|tile_d0=32
                   128,                                   // tile_dim1=128 (tile_dim2=0)
                   rowStride,                             // tensor_dim0_stride[31:0]
                   0, 0 };
    i32x4_t gz  = { 0, 0, 0, 0 };
    i32x8_t gz8 = { 0, 0, 0, 0, 0, 0, 0, 0 };
#if __clang_major__ >= 23
    __builtin_amdgcn_tensor_load_to_lds(g0, g1, gz, gz, gz8, 0);
#else
    __builtin_amdgcn_tensor_load_to_lds(g0, g1, gz, gz, 0);
    (void)gz8;
#endif
  };
#endif

  // ---- tile staging ----
  auto stageA = [&](int buf, int k0) {
#if HAS_TDM
    if (wave == 0)       // A[m0..m0+127][k0..k0+31]
      tdm_tile(A + (size_t)m0 * K + k0, &As[buf][0][0], K, M_);
#else
    const uint4* s = (const uint4*)(A + (size_t)(m0 + trow) * K + k0 + tcol);
    uint4* d = (uint4*)&As[buf][trow][tcol];
    d[0] = s[0]; d[1] = s[1];
#endif
  };
  auto stageB = [&](int buf, int k0) {
#if HAS_TDM
    if (wave == 1)       // Bt[n0..n0+127][k0..k0+31]
      tdm_tile(Bt + (size_t)n0 * K + k0, &Bs[buf][0][0], K, N);
#else
    const uint4* s = (const uint4*)(Bt + (size_t)(n0 + trow) * K + k0 + tcol);
    uint4* d = (uint4*)&Bs[buf][trow][tcol];
    d[0] = s[0]; d[1] = s[1];
#endif
  };
  auto waitStage = [&]() {
#if HAS_TDM
    __builtin_amdgcn_s_wait_tensorcnt(0);   // no-op for waves that issued none
#endif
  };

  // ---- pipelined main loop ----
  stageA(0, 0);
  stageB(0, 0);
  waitStage();
  __syncthreads();

  int buf = 0;
  for (int k0 = 0; k0 < K; k0 += 32) {
    const int nk = k0 + 32;
    if (nk < K) {                        // stage next tile into other buffer
      stageA(buf ^ 1, nk);
      stageB(buf ^ 1, nk);
      if (nk + 32 < K) {                 // keep a prefetch ahead of the DMA
        __builtin_prefetch(A  + (size_t)(m0 + trow) * K + nk + 32 + tcol, 0, 3);
        __builtin_prefetch(Bt + (size_t)(n0 + trow) * K + nk + 32 + tcol, 0, 3);
      }
    }

    // ---- fragments per ISA 16-bit A/B layout (wave32) ----
    union Fr { v16h h; uint32_t u[8]; };
    Fr a[2], b[4];
    const int lm = lane & 15;
    const int kb = (lane >> 4) * 8;      // lanes 0-15: K base 0; 16-31: K base 8
#pragma unroll
    for (int mt = 0; mt < 2; ++mt) {
      const int row = wm * 32 + mt * 16 + lm;
#pragma unroll
      for (int j = 0; j < 8; ++j) {
        const int kk = ((j >> 2) * 16) + kb + (j & 3) * 2;
        a[mt].u[j] = *(const uint32_t*)&As[buf][row][kk];
      }
    }
#pragma unroll
    for (int nt = 0; nt < 4; ++nt) {
      const int col = wn * 64 + nt * 16 + lm;
#pragma unroll
      for (int j = 0; j < 8; ++j) {
        const int kk = ((j >> 2) * 16) + kb + (j & 3) * 2;
        b[nt].u[j] = *(const uint32_t*)&Bs[buf][col][kk];
      }
    }
#pragma unroll
    for (int mt = 0; mt < 2; ++mt)
#pragma unroll
      for (int nt = 0; nt < 4; ++nt)
        acc[mt][nt] = __builtin_amdgcn_wmma_f32_16x16x32_f16(
            false, a[mt].h, false, b[nt].h, (short)0, acc[mt][nt], false, false);

    waitStage();                         // next tile fully in LDS
    __syncthreads();
    buf ^= 1;
  }

  // ---- epilogue: bias, SiLU, residual, f32/f16 store ----
#pragma unroll
  for (int mt = 0; mt < 2; ++mt) {
#pragma unroll
    for (int nt = 0; nt < 4; ++nt) {
      const int n = n0 + wn * 64 + nt * 16 + (lane & 15);
#pragma unroll
      for (int r = 0; r < 8; ++r) {
        const int m = m0 + wm * 32 + mt * 16 + r + ((lane >> 4) << 3);
        float v = acc[mt][nt][r];
        if (bias) v += bias[n];
        if (act == 1) v = v / (1.f + __expf(-v));          // SiLU
        if (res)  v = res[(size_t)m * N + n] + resScale * v;
        if (o32)  o32[(size_t)m * N + n] = v;
        if (o16)  o16[(size_t)m * N + n] = (_Float16)v;
      }
    }
  }
}

// =====================================================================
// Windowed MHSA: one thread per (b,h,t); two-pass online softmax over
// s in [t-32, t+32]. qkv f16 [8192,1536] laid out [3,H,DH]; out f16 [8192,512].
// =====================================================================
__global__ __launch_bounds__(256) void k_attn(const _Float16* __restrict__ qkv,
                                              _Float16* __restrict__ out) {
  int idx = blockIdx.x * 256 + threadIdx.x;   // B*H*T = 65536 exactly
  int t = idx & (T_ - 1);
  int h = (idx >> 10) & (H_ - 1);
  int b = idx >> 13;
  const _Float16* base = qkv + (size_t)(b * T_) * (3 * D_);
  const _Float16* qrow = base + (size_t)t * (3 * D_) + h * DH_;
  float q[DH_];
#pragma unroll
  for (int d = 0; d < DH_; ++d) q[d] = (float)qrow[d];
  int s0 = (t - 32 > 0) ? t - 32 : 0;
  int s1 = (t + 32 < T_ - 1) ? t + 32 : T_ - 1;
  float mmax = -3.4e38f, lsum = 0.f;
  for (int s = s0; s <= s1; ++s) {
    const _Float16* kr = base + (size_t)s * (3 * D_) + D_ + h * DH_;
    float dot = 0.f;
#pragma unroll
    for (int d = 0; d < DH_; ++d) dot += q[d] * (float)kr[d];
    dot *= 0.125f;   // DH^-0.5
    if (dot > mmax) { lsum = lsum * __expf(mmax - dot) + 1.f; mmax = dot; }
    else            { lsum += __expf(dot - mmax); }
  }
  float o[DH_];
#pragma unroll
  for (int d = 0; d < DH_; ++d) o[d] = 0.f;
  for (int s = s0; s <= s1; ++s) {
    const _Float16* kr = base + (size_t)s * (3 * D_) + D_ + h * DH_;
    const _Float16* vr = base + (size_t)s * (3 * D_) + 2 * D_ + h * DH_;
    float dot = 0.f;
#pragma unroll
    for (int d = 0; d < DH_; ++d) dot += q[d] * (float)kr[d];
    float wgt = __expf(dot * 0.125f - mmax);
#pragma unroll
    for (int d = 0; d < DH_; ++d) o[d] += wgt * (float)vr[d];
  }
  float inv = 1.f / lsum;
  _Float16* orow = out + (size_t)(b * T_ + t) * D_ + h * DH_;
#pragma unroll
  for (int d = 0; d < DH_; ++d) orow[d] = (_Float16)(o[d] * inv);
}

// =====================================================================
// GLU: in [M,2*DC] f16 -> out [M,DC] f16, out = a1 * sigmoid(a2)
// =====================================================================
__global__ __launch_bounds__(256) void k_glu(const _Float16* __restrict__ x,
                                             _Float16* __restrict__ out) {
  int e = blockIdx.x * 256 + threadIdx.x;   // M_*DC_
  int c = e & (DC_ - 1);
  int m = e >> 10;
  float a1 = (float)x[(size_t)m * (2 * DC_) + c];
  float a2 = (float)x[(size_t)m * (2 * DC_) + DC_ + c];
  out[e] = (_Float16)(a1 / (1.f + __expf(-a2)));
}

// =====================================================================
// Depthwise conv along T, K=31, pad 15, per-channel filter dw_w[c][0][j]
// =====================================================================
__global__ __launch_bounds__(256) void k_dwconv(const _Float16* __restrict__ x,
                                                const float* __restrict__ w,
                                                const float* __restrict__ bias,
                                                _Float16* __restrict__ out) {
  int e = blockIdx.x * 256 + threadIdx.x;   // M_*DC_
  int c = e & (DC_ - 1);
  int m = e >> 10;
  int t = m & (T_ - 1);
  int b = m >> 10;
  float acc = bias[c];
  const float* wc = w + c * KW_;
#pragma unroll
  for (int j = 0; j < KW_; ++j) {
    int ts = t + j - KW_ / 2;
    if (ts >= 0 && ts < T_)
      acc += (float)x[((size_t)(b * T_ + ts) << 10) + c] * wc[j];
  }
  out[e] = (_Float16)acc;
}

// =====================================================================
// GroupNorm(groups=1) over (C,T) per batch: two-stage (atomic sums) + apply+SiLU
// =====================================================================
__global__ void k_zero16(float* p) { if (threadIdx.x < 16) p[threadIdx.x] = 0.f; }

__global__ __launch_bounds__(256) void k_gn_reduce(const _Float16* __restrict__ x,
                                                   float* __restrict__ sums) {
  int e = blockIdx.x * 256 + threadIdx.x;   // M_*DC_; 32-lane groups never cross batch
  float v = (float)x[e];
  float s = v, ss = v * v;
  for (int off = 16; off; off >>= 1) {
    s  += __shfl_down(s,  off, 32);
    ss += __shfl_down(ss, off, 32);
  }
  if ((threadIdx.x & 31) == 0) {
    int b = e >> 20;                         // e / (T_*DC_)
    atomicAdd(&sums[b * 2 + 0], s);
    atomicAdd(&sums[b * 2 + 1], ss);
  }
}

__global__ __launch_bounds__(256) void k_gn_apply(const _Float16* __restrict__ x,
                                                  const float* __restrict__ sums,
                                                  const float* __restrict__ gw,
                                                  const float* __restrict__ gb,
                                                  _Float16* __restrict__ out) {
  int e = blockIdx.x * 256 + threadIdx.x;
  int c = e & (DC_ - 1);
  int b = e >> 20;
  const float inv = 1.f / (float)(DC_ * T_);
  float mean = sums[b * 2 + 0] * inv;
  float var  = sums[b * 2 + 1] * inv - mean * mean;
  float rstd = rsqrtf(var + EPS_);
  float v = ((float)x[e] - mean) * rstd * gw[c] + gb[c];
  v = v / (1.f + __expf(-v));                // SiLU
  out[e] = (_Float16)v;
}

// =====================================================================
// launcher
// =====================================================================
extern "C" void kernel_launch(void* const* d_in, const int* in_sizes, int n_in,
                              void* d_out, int out_size, void* d_ws, size_t ws_size,
                              hipStream_t stream) {
  (void)in_sizes; (void)n_in; (void)out_size; (void)ws_size;

  // input dict order
  const float* x        = (const float*)d_in[0];
  const float* f1_ln_w  = (const float*)d_in[1];
  const float* f1_ln_b  = (const float*)d_in[2];
  const float* f1_w1    = (const float*)d_in[3];
  const float* f1_b1    = (const float*)d_in[4];
  const float* f1_w2    = (const float*)d_in[5];
  const float* f1_b2    = (const float*)d_in[6];
  const float* at_ln_w  = (const float*)d_in[7];
  const float* at_ln_b  = (const float*)d_in[8];
  const float* qkv_w    = (const float*)d_in[9];
  const float* qkv_b    = (const float*)d_in[10];
  const float* out_w    = (const float*)d_in[11];
  const float* out_b    = (const float*)d_in[12];
  const float* cv_ln_w  = (const float*)d_in[13];
  const float* cv_ln_b  = (const float*)d_in[14];
  const float* pw1_w    = (const float*)d_in[15];
  const float* pw1_b    = (const float*)d_in[16];
  const float* dw_w     = (const float*)d_in[17];
  const float* dw_b     = (const float*)d_in[18];
  const float* gn_w     = (const float*)d_in[19];
  const float* gn_b     = (const float*)d_in[20];
  const float* pw2_w    = (const float*)d_in[21];
  const float* pw2_b    = (const float*)d_in[22];
  const float* f2_ln_w  = (const float*)d_in[23];
  const float* f2_ln_b  = (const float*)d_in[24];
  const float* f2_w1    = (const float*)d_in[25];
  const float* f2_b1    = (const float*)d_in[26];
  const float* f2_w2    = (const float*)d_in[27];
  const float* f2_b2    = (const float*)d_in[28];
  const float* fn_ln_w  = (const float*)d_in[29];
  const float* fn_ln_b  = (const float*)d_in[30];

  // ---- workspace carving (all weights stored transposed: [N][K] f16) ----
  uint8_t* wsb = (uint8_t*)d_ws;
  size_t off = 0;
  auto alloc = [&](size_t bytes) -> void* {
    void* p = wsb + off;
    off += (bytes + 255) & ~(size_t)255;
    return p;
  };
  _Float16* w_f1w1 = (_Float16*)alloc((size_t)2048 * 512 * 2);   // [DF][D]
  _Float16* w_f1w2 = (_Float16*)alloc((size_t)512 * 2048 * 2);   // [D][DF]
  _Float16* w_qkv  = (_Float16*)alloc((size_t)1536 * 512 * 2);   // [3D][D]
  _Float16* w_out  = (_Float16*)alloc((size_t)512 * 512 * 2);    // [D][D]
  _Float16* w_pw1  = (_Float16*)alloc((size_t)2048 * 512 * 2);   // [2DC][D] (native)
  _Float16* w_pw2  = (_Float16*)alloc((size_t)512 * 1024 * 2);   // [D][DC] (native)
  _Float16* w_f2w1 = (_Float16*)alloc((size_t)2048 * 512 * 2);   // [DF][D]
  _Float16* w_f2w2 = (_Float16*)alloc((size_t)512 * 2048 * 2);   // [D][DF]
  float*    x_cur  = (float*)   alloc((size_t)M_ * D_ * 4);      // residual stream
  _Float16* bufA   = (_Float16*)alloc((size_t)M_ * DF_ * 2);     // generic f16 buf
  _Float16* bufB   = (_Float16*)alloc((size_t)M_ * DF_ * 2);     // generic f16 buf
  float*    gn_s   = (float*)   alloc(64);

  auto cvt = [&](const float* s, _Float16* d, int n) {
    k_f32_to_f16<<<(n + 255) / 256, 256, 0, stream>>>(s, d, n);
  };
  auto cvtT = [&](const float* s, _Float16* d, int R, int C) {   // [R][C] -> [C][R]
    k_transpose_f32_to_f16<<<(R * C + 255) / 256, 256, 0, stream>>>(s, d, R, C);
  };
  auto gemm = [&](const _Float16* A, const _Float16* Bt, const float* bias,
                  float* o32, _Float16* o16, const float* res, float rs,
                  int N, int K, int act) {
    dim3 g(N / 128, M_ / 128);
    k_gemm_wmma<<<g, 256, 0, stream>>>(A, Bt, bias, o32, o16, res, rs, N, K, act);
  };
  auto ln = [&](const float* w, const float* b, _Float16* o16, float* o32) {
    k_layernorm<<<M_, 256, 0, stream>>>(x_cur, w, b, o16, o32);
  };

  const int EW = M_ * DC_;                 // 8388608 elementwise work
  const int EG = EW / 256;                 // grid for elementwise kernels

  // ---- weight prep: every GEMM B-operand becomes [N][K] f16 ----
  cvtT(f1_w1, w_f1w1, 512, 2048);          // (D,DF)  -> [DF][D]
  cvtT(f1_w2, w_f1w2, 2048, 512);          // (DF,D)  -> [D][DF]
  cvtT(qkv_w, w_qkv, 512, 1536);           // (D,3D)  -> [3D][D]
  cvtT(out_w, w_out, 512, 512);            // (D,D)   -> [D][D]
  cvt(pw1_w, w_pw1, 2048 * 512);           // (2DC,D) already [N][K]
  cvt(pw2_w, w_pw2, 512 * 1024);           // (D,DC)  already [N][K]
  cvtT(f2_w1, w_f2w1, 512, 2048);
  cvtT(f2_w2, w_f2w2, 2048, 512);

  // residual stream init (don't mutate inputs)
  (void)hipMemcpyAsync(x_cur, x, (size_t)M_ * D_ * 4, hipMemcpyDeviceToDevice, stream);

  // ---- macaron FFN1: x += 0.5 * W2(silu(W1(ln(x)))) ----
  ln(f1_ln_w, f1_ln_b, bufA, nullptr);
  gemm(bufA, w_f1w1, f1_b1, nullptr, bufB, nullptr, 0.f, DF_, D_, /*silu*/1);
  gemm(bufB, w_f1w2, f1_b2, x_cur, nullptr, x_cur, 0.5f, D_, DF_, 0);

  // ---- windowed MHSA ----
  ln(at_ln_w, at_ln_b, bufA, nullptr);
  gemm(bufA, w_qkv, qkv_b, nullptr, bufB, nullptr, 0.f, 3 * D_, D_, 0);
  k_attn<<<(B_ * H_ * T_) / 256, 256, 0, stream>>>(bufB, bufA);
  gemm(bufA, w_out, out_b, x_cur, nullptr, x_cur, 1.0f, D_, D_, 0);

  // ---- conv module ----
  ln(cv_ln_w, cv_ln_b, bufA, nullptr);
  gemm(bufA, w_pw1, pw1_b, nullptr, bufB, nullptr, 0.f, 2 * DC_, D_, 0);
  k_glu<<<EG, 256, 0, stream>>>(bufB, bufA);                  // [M,DC]
  k_dwconv<<<EG, 256, 0, stream>>>(bufA, dw_w, dw_b, bufB);   // [M,DC]
  k_zero16<<<1, 16, 0, stream>>>(gn_s);
  k_gn_reduce<<<EG, 256, 0, stream>>>(bufB, gn_s);
  k_gn_apply<<<EG, 256, 0, stream>>>(bufB, gn_s, gn_w, gn_b, bufA);
  gemm(bufA, w_pw2, pw2_b, x_cur, nullptr, x_cur, 1.0f, D_, DC_, 0);

  // ---- macaron FFN2 ----
  ln(f2_ln_w, f2_ln_b, bufA, nullptr);
  gemm(bufA, w_f2w1, f2_b1, nullptr, bufB, nullptr, 0.f, DF_, D_, /*silu*/1);
  gemm(bufB, w_f2w2, f2_b2, x_cur, nullptr, x_cur, 0.5f, D_, DF_, 0);

  // ---- final LN -> f32 output ----
  ln(fn_ln_w, fn_ln_b, nullptr, (float*)d_out);
}
